// LlamaAttentionFused_13718125543560
// MI455X (gfx1250) — matrix operand order repaired and checked by reading, not compiled, not run on hardware
//
#include <hip/hip_runtime.h>
#include <hip/hip_bf16.h>

// Problem constants
constexpr int Bsz = 2;
constexpr int Ssz = 2048;
constexpr int Dsz = 4096;
constexpr int Hn  = 32;
constexpr int HDs = 128;   // head dim

typedef __attribute__((ext_vector_type(16))) __bf16 bf16x16;
typedef __attribute__((ext_vector_type(2)))  __bf16 bf16x2;
typedef __attribute__((ext_vector_type(8)))  float  f32x8;

union AFrag { bf16x16 v; uint4 q[2]; };
union CFrag { f32x8  v; float f[8]; };

// Native bf16 conversions (hardware cvt, RTNE)
__device__ __forceinline__ unsigned short f2bf(float f) {
  union { __bf16 h; unsigned short u; } cv;
  cv.h = (__bf16)f;
  return cv.u;
}
__device__ __forceinline__ unsigned pkbf2(float a, float b) {
  union { bf16x2 h; unsigned u; } cv;
  cv.h.x = (__bf16)a;
  cv.h.y = (__bf16)b;
  return cv.u;
}
__device__ __forceinline__ float bf2f(unsigned short h) {
  return __uint_as_float(((unsigned)h) << 16);
}

// CDNA5 async global->LDS copy (ASYNCcnt path), GV addressing (64-bit vaddr).
__device__ __forceinline__ void async_g2l_b128(const void* gptr, void* lptr) {
  unsigned lds = (unsigned)(size_t)lptr;                 // flat->LDS: addr[31:0]
  unsigned long long ga = (unsigned long long)(size_t)gptr;
  asm volatile("global_load_async_to_lds_b128 %0, %1, off"
               :: "v"(lds), "v"(ga) : "memory");
}
__device__ __forceinline__ void wait_asynccnt0() {
  asm volatile("s_wait_asynccnt 0x0" ::: "memory");
}

// ---------------------------------------------------------------------------
// Tiled GEMM: C[M,N] = A[M,K] * W[K,N], bf16 WMMA, f32 accumulate.
// BM=128, BN=128, BK=64; 256 threads = 8 wave32s (4 rows x 2 cols);
// each wave computes a 32x64 sub-tile = 2x4 WMMA tiles; 16 WMMAs/k-step.
// ---------------------------------------------------------------------------
#define GBM 128
#define GBN 128
#define GBK 64

template<bool A_BF16, bool OUT_F32>
__global__ __launch_bounds__(256) void gemm_bf16_wmma(
    const void* __restrict__ Av, const float* __restrict__ W,
    void* __restrict__ Cv, int M, int N, int K)
{
  __shared__ unsigned short lA[GBM][GBK];   // [row][k]          16 KB
  __shared__ unsigned short lB[GBN][GBK];   // [n][k] transposed 16 KB

  const int tid  = threadIdx.x;
  const int wave = tid >> 5;
  const int lane = tid & 31;
  const int half = lane >> 4;   // 0: lanes 0-15, 1: lanes 16-31
  const int l16  = lane & 15;

  const int wm = (wave & 3) * 32;   // wave row offset inside block tile
  const int wn = (wave >> 2) * 64;  // wave col offset inside block tile

  const int m0 = blockIdx.y * GBM;
  const int n0 = blockIdx.x * GBN;

  CFrag acc[2][4];
#pragma unroll
  for (int i = 0; i < 2; i++)
#pragma unroll
    for (int j = 0; j < 4; j++)
#pragma unroll
      for (int r = 0; r < 8; r++) acc[i][j].f[r] = 0.0f;

  const float*          Af = (const float*)Av;
  const unsigned short* Ab = (const unsigned short*)Av;

  for (int k0 = 0; k0 < K; k0 += GBK) {
    // ---- stage A tile (128x64) into LDS as bf16 ----
    if (A_BF16) {
      // already bf16: pure copy -> async global->LDS (ASYNCcnt path)
      // 8192 bf16 = 1024 b128 slots; 4 per thread
#pragma unroll
      for (int i = 0; i < 4; i++) {
        int s   = tid + i * 256;      // 0..1023
        int row = s >> 3;             // 8 b128 per 64-elem row
        int c8  = s & 7;
        async_g2l_b128(Ab + (size_t)(m0 + row) * K + k0 + c8 * 8,
                       &lA[row][c8 * 8]);
      }
    } else {
      // fp32 -> bf16 with packed converts; 2048 float4 slots; 8 per thread
#pragma unroll
      for (int i = 0; i < 8; i++) {
        int s   = tid + i * 256;      // 0..2047
        int row = s >> 4;             // 16 float4 per row
        int c4  = s & 15;
        float4 d = *(const float4*)(Af + (size_t)(m0 + row) * K + k0 + c4 * 4);
        uint2 st;
        st.x = pkbf2(d.x, d.y);
        st.y = pkbf2(d.z, d.w);
        *(uint2*)&lA[row][c4 * 4] = st;
      }
    }
    // ---- stage W tile (64x128) into LDS transposed [n][k] ----
    // read two consecutive k-rows, pack (k,k+1) pairs -> dword LDS stores
#pragma unroll
    for (int i = 0; i < 4; i++) {
      int s  = tid + i * 256;   // 0..1023
      int kp = s >> 5;          // k-pair index 0..31
      int n4 = s & 31;          // float4 group along n
      const float* wp = W + (size_t)(k0 + 2 * kp) * N + n0 + n4 * 4;
      float4 d0 = *(const float4*)wp;
      float4 d1 = *(const float4*)(wp + N);
      const float* p0 = (const float*)&d0;
      const float* p1 = (const float*)&d1;
#pragma unroll
      for (int j = 0; j < 4; j++)
        *(unsigned*)&lB[n4 * 4 + j][kp * 2] = pkbf2(p0[j], p1[j]);
    }
    if (A_BF16) wait_asynccnt0();
    __syncthreads();

    // ---- compute: 2 k-steps of 32, 16 WMMAs total ----
#pragma unroll
    for (int ks = 0; ks < 2; ks++) {
      AFrag a[2], b[4];
#pragma unroll
      for (int i = 0; i < 2; i++) {
        // A 16x32: lane<16 -> row=l16, K {0..7,16..23}; lane>=16 -> {8..15,24..31}
        const unsigned short* p = &lA[wm + i * 16 + l16][ks * 32 + half * 8];
        a[i].q[0] = *(const uint4*)p;
        a[i].q[1] = *(const uint4*)(p + 16);
      }
#pragma unroll
      for (int j = 0; j < 4; j++) {
        // B 32x16: lane<16 -> col=l16, K {0..7,16..23}; lane>=16 -> {8..15,24..31}
        const unsigned short* p = &lB[wn + j * 16 + l16][ks * 32 + half * 8];
        b[j].q[0] = *(const uint4*)p;
        b[j].q[1] = *(const uint4*)(p + 16);
      }
#pragma unroll
      for (int i = 0; i < 2; i++)
#pragma unroll
        for (int j = 0; j < 4; j++)
          acc[i][j].v = __builtin_amdgcn_wmma_f32_16x16x32_bf16(
              false, a[i].v, false, b[j].v, (short)0, acc[i][j].v, false, false);
    }
    __syncthreads();
  }

  // ---- writeback: C layout VGPR r -> row r + half*8, col = l16 ----
#pragma unroll
  for (int i = 0; i < 2; i++) {
#pragma unroll
    for (int j = 0; j < 4; j++) {
      int col = n0 + wn + j * 16 + l16;
#pragma unroll
      for (int r = 0; r < 8; r++) {
        int row = m0 + wm + i * 16 + r + half * 8;
        float v = acc[i][j].f[r];
        if (OUT_F32) ((float*)Cv)[(size_t)row * N + col] = v;
        else ((unsigned short*)Cv)[(size_t)row * N + col] = f2bf(v);
      }
    }
  }
}

// ---------------------------------------------------------------------------
// RoPE on Q and K (bf16, layout [B,S,H,HD], pairs (d, d+64) within a head)
// ---------------------------------------------------------------------------
__global__ void rope_kernel(unsigned short* __restrict__ Q,
                            unsigned short* __restrict__ Kt,
                            const float* __restrict__ fcos,
                            const float* __restrict__ fsin)
{
  size_t idx = (size_t)blockIdx.x * blockDim.x + threadIdx.x;
  const size_t total = (size_t)Bsz * Ssz * Hn * 64;
  if (idx >= total) return;
  int d      = idx & 63;
  size_t t   = idx >> 6;
  int h      = t & (Hn - 1);
  size_t bs  = t >> 5;            // b*S + s
  int s      = (int)(bs & (Ssz - 1));
  float c  = fcos[s * 64 + d];
  float sn = fsin[s * 64 + d];
  size_t i1 = bs * Dsz + (size_t)h * HDs + d;
  size_t i2 = i1 + 64;
  float q1 = bf2f(Q[i1]), q2 = bf2f(Q[i2]);
  Q[i1] = f2bf(q1 * c - q2 * sn);
  Q[i2] = f2bf(q1 * sn + q2 * c);
  float k1 = bf2f(Kt[i1]), k2 = bf2f(Kt[i2]);
  Kt[i1] = f2bf(k1 * c - k2 * sn);
  Kt[i2] = f2bf(k1 * sn + k2 * c);
}

// ---------------------------------------------------------------------------
// Flash attention: grid (S/128, H, B); 256 threads = 8 waves; each wave owns
// 16 query rows; K/V streamed in 32-key blocks through LDS; online softmax.
// K block staged with async global->LDS; V staged transposed via VGPRs.
// ---------------------------------------------------------------------------
#define FQB 128
#define FKB 32

__global__ __launch_bounds__(256) void flash_attn(
    const unsigned short* __restrict__ Q,
    const unsigned short* __restrict__ Kt,
    const unsigned short* __restrict__ V,
    unsigned short* __restrict__ Ao)
{
  __shared__ unsigned short lK[FKB][HDs];    // [key][dim]          8 KB
  __shared__ unsigned short lV[HDs][FKB];    // [dim][key] (transp) 8 KB
  __shared__ unsigned short lP[8][16][FKB];  // per-wave P scratch  8 KB

  const int tid  = threadIdx.x;
  const int wave = tid >> 5;
  const int lane = tid & 31;
  const int half = lane >> 4;
  const int l16  = lane & 15;

  const int qb = blockIdx.x;
  const int h  = blockIdx.y;
  const int b  = blockIdx.z;
  const int q_lo = qb * FQB;

  // Q fragments: 16 rows x 128 dims = 4 A-fragments, kept in registers
  const size_t rowQ = (size_t)b * Ssz + q_lo + wave * 16 + l16;
  const unsigned short* qptr = Q + rowQ * Dsz + (size_t)h * HDs;
  AFrag qf[4];
#pragma unroll
  for (int f = 0; f < 4; f++) {
    qf[f].q[0] = *(const uint4*)(qptr + f * 32 + half * 8);
    qf[f].q[1] = *(const uint4*)(qptr + f * 32 + half * 8 + 16);
  }

  CFrag o[8];
#pragma unroll
  for (int ot = 0; ot < 8; ot++)
#pragma unroll
    for (int r = 0; r < 8; r++) o[ot].f[r] = 0.0f;
  float mrun[8], lrun[8];
#pragma unroll
  for (int r = 0; r < 8; r++) { mrun[r] = -1e30f; lrun[r] = 0.0f; }

  const float scale = 0.08838834764831845f;  // 1/sqrt(128)
  const int nkb = qb * 4 + 4;                // causal bound on key blocks

  for (int kb = 0; kb < nkb; kb++) {
    // ---- stage K (async, row-major) and V (transposed) into LDS ----
#pragma unroll
    for (int i = 0; i < 2; i++) {
      int sIdx = tid + i * 256;   // 0..511
      int key  = sIdx >> 4;       // 16 b128 per 128-dim row
      int d8   = sIdx & 15;
      size_t grow = ((size_t)b * Ssz + kb * FKB + key) * Dsz + (size_t)h * HDs + d8 * 8;
      async_g2l_b128(Kt + grow, &lK[key][d8 * 8]);      // ASYNCcnt path
      uint4 vv = *(const uint4*)(V + grow);
      const unsigned short* pv = (const unsigned short*)&vv;
#pragma unroll
      for (int j = 0; j < 8; j++) lV[d8 * 8 + j][key] = pv[j];
    }
    wait_asynccnt0();
    __syncthreads();

    // ---- scores S = Q @ K^T (16x32), two 16x16 col-tiles ----
    CFrag sc[2];
#pragma unroll
    for (int ct = 0; ct < 2; ct++) {
#pragma unroll
      for (int r = 0; r < 8; r++) sc[ct].f[r] = 0.0f;
#pragma unroll
      for (int f = 0; f < 4; f++) {
        AFrag bf_;
        const unsigned short* p = &lK[ct * 16 + l16][f * 32 + half * 8];
        bf_.q[0] = *(const uint4*)p;
        bf_.q[1] = *(const uint4*)(p + 16);
        sc[ct].v = __builtin_amdgcn_wmma_f32_16x16x32_bf16(
            false, qf[f].v, false, bf_.v, (short)0, sc[ct].v, false, false);
      }
    }

    // ---- online softmax (rows live across the 16 lanes of one half) ----
    const int qrow_base = q_lo + wave * 16;
#pragma unroll
    for (int r = 0; r < 8; r++) {
      int qi = qrow_base + r + half * 8;
      float mx = -1e30f;
#pragma unroll
      for (int ct = 0; ct < 2; ct++) {
        int ki = kb * FKB + ct * 16 + l16;
        float sv = sc[ct].f[r] * scale + (ki <= qi ? 0.0f : -1e9f);
        sc[ct].f[r] = sv;
        mx = fmaxf(mx, sv);
      }
#pragma unroll
      for (int off = 1; off < 16; off <<= 1)
        mx = fmaxf(mx, __shfl_xor(mx, off, 32));
      float mnew  = fmaxf(mrun[r], mx);
      float alpha = __expf(mrun[r] - mnew);
      mrun[r] = mnew;
      float rs = 0.0f;
#pragma unroll
      for (int ct = 0; ct < 2; ct++) {
        float p = __expf(sc[ct].f[r] - mnew);
        sc[ct].f[r] = p;
        rs += p;
      }
#pragma unroll
      for (int off = 1; off < 16; off <<= 1)
        rs += __shfl_xor(rs, off, 32);
      lrun[r] = lrun[r] * alpha + rs;
#pragma unroll
      for (int ot = 0; ot < 8; ot++) o[ot].f[r] *= alpha;
    }

    // ---- transpose P (C-layout -> A-layout) via per-wave LDS scratch ----
#pragma unroll
    for (int ct = 0; ct < 2; ct++)
#pragma unroll
      for (int r = 0; r < 8; r++)
        lP[wave][r + half * 8][ct * 16 + l16] = f2bf(sc[ct].f[r]);
    AFrag pf;   // same-wave LDS dependency; compiler emits s_wait_dscnt
    {
      const unsigned short* p = &lP[wave][l16][half * 8];
      pf.q[0] = *(const uint4*)p;
      pf.q[1] = *(const uint4*)(p + 16);
    }

    // ---- O += P @ V : 8 dim-tiles of 16x16 ----
#pragma unroll
    for (int ot = 0; ot < 8; ot++) {
      AFrag bf_;
      const unsigned short* p = &lV[ot * 16 + l16][half * 8];
      bf_.q[0] = *(const uint4*)p;
      bf_.q[1] = *(const uint4*)(p + 16);
      o[ot].v = __builtin_amdgcn_wmma_f32_16x16x32_bf16(
          false, pf.v, false, bf_.v, (short)0, o[ot].v, false, false);
    }
    __syncthreads();
  }

  // ---- normalize and write attention output (bf16) ----
#pragma unroll
  for (int r = 0; r < 8; r++) {
    float inv = 1.0f / lrun[r];
    size_t row = (size_t)b * Ssz + q_lo + wave * 16 + r + half * 8;
#pragma unroll
    for (int ot = 0; ot < 8; ot++)
      Ao[row * Dsz + (size_t)h * HDs + ot * 16 + l16] = f2bf(o[ot].f[r] * inv);
  }
}

// ---------------------------------------------------------------------------
extern "C" void kernel_launch(void* const* d_in, const int* in_sizes, int n_in,
                              void* d_out, int out_size, void* d_ws, size_t ws_size,
                              hipStream_t stream) {
  const float* x    = (const float*)d_in[0];
  const float* fcos = (const float*)d_in[1];
  const float* fsin = (const float*)d_in[2];
  // d_in[3] = mask (unused; causal mask computed analytically)
  const float* wq   = (const float*)d_in[4];
  const float* wk   = (const float*)d_in[5];
  const float* wv   = (const float*)d_in[6];
  const float* wo   = (const float*)d_in[7];
  float* out = (float*)d_out;

  const size_t TN = (size_t)Bsz * Ssz * Dsz;   // 16M elements
  unsigned short* Qb = (unsigned short*)d_ws;
  unsigned short* Kb = Qb + TN;
  unsigned short* Vb = Kb + TN;
  unsigned short* Ab = Vb + TN;

  const int M = Bsz * Ssz;
  dim3 gg(Dsz / GBN, M / GBM);

  gemm_bf16_wmma<false, false><<<gg, 256, 0, stream>>>(x, wq, Qb, M, Dsz, Dsz);
  gemm_bf16_wmma<false, false><<<gg, 256, 0, stream>>>(x, wk, Kb, M, Dsz, Dsz);
  gemm_bf16_wmma<false, false><<<gg, 256, 0, stream>>>(x, wv, Vb, M, Dsz, Dsz);

  size_t pairs = (size_t)Bsz * Ssz * Hn * 64;
  rope_kernel<<<dim3((unsigned)((pairs + 255) / 256)), 256, 0, stream>>>(Qb, Kb, fcos, fsin);

  flash_attn<<<dim3(Ssz / FQB, Hn, Bsz), 256, 0, stream>>>(Qb, Kb, Vb, Ab);

  gemm_bf16_wmma<true, true><<<gg, 256, 0, stream>>>(Ab, wo, out, M, Dsz, Dsz);
}
